// Transformer_38293928411348
// MI455X (gfx1250) — matrix operand order repaired and checked by reading, not compiled, not run on hardware
//
#include <hip/hip_runtime.h>
#include <hip/hip_bf16.h>

// ---------------------------------------------------------------------------
// CDNA5 / gfx1250 transformer forward. All GEMMs use v_wmma_f32_16x16x32_bf16
// (wave32, A 16x32 bf16 frag = 2x16B loads/lane; B 32x16 frag = 1x32B load/lane
// k = 16*(lane/16)+s order; attn_ctx uses ISA table order). fp32 accumulate throughout.
// ---------------------------------------------------------------------------

typedef __bf16 bf16;
typedef __bf16 v16bf  __attribute__((ext_vector_type(16)));
typedef __bf16 bf16x8 __attribute__((ext_vector_type(8)));
typedef float  v8f    __attribute__((ext_vector_type(8)));

#define B_      8
#define S_      512
#define EMB_    512
#define HEADS_  8
#define DK_     64
#define FF_     2048
#define VOCAB_  32000
#define NROWS   (B_ * S_)            // 4096 rows for both enc and dec

// ---------------------------------------------------------------------------
// Generic GEMM: C[M,N] = A[M,K](bf16) @ W[N,K]^T(bf16) + bias, optional relu.
// Block = 256 thr (8 waves, 2x4). Wave tile = 32x64 (2 M-tiles x 4 N-tiles =
// 8 WMMA per K-step, A fragments reused 4x). Block tile = 64x256.
// Requires M%64==0, N%256==0, K%32==0 (true for all call sites).
// ---------------------------------------------------------------------------
__global__ __launch_bounds__(256) void gemm_bf16_nt(
    const bf16* __restrict__ A, int ldA,
    const bf16* __restrict__ W, int ldW,
    float* __restrict__ Cf, bf16* __restrict__ Ch, int ldC,
    const float* __restrict__ bias, int relu, int K)
{
    const int lane  = threadIdx.x & 31;
    const int wave  = threadIdx.x >> 5;
    const int waveM = wave & 1, waveN = wave >> 1;
    const int m0 = blockIdx.x * 64  + waveM * 32;
    const int n0 = blockIdx.y * 256 + waveN * 64;
    const int l15 = lane & 15, g = lane >> 4;

    v8f acc[2][4] = {};
    const bf16* arow[2];
    const bf16* brow[4];
    #pragma unroll
    for (int mt = 0; mt < 2; ++mt) arow[mt] = A + (size_t)(m0 + mt * 16 + l15) * ldA;
    #pragma unroll
    for (int nt = 0; nt < 4; ++nt) brow[nt] = W + (size_t)(n0 + nt * 16 + l15) * ldW;

    for (int kk = 0; kk < K; kk += 32) {
        v16bf af[2], bfr[4];
        #pragma unroll
        for (int mt = 0; mt < 2; ++mt) {
            union { v16bf v; bf16x8 hh[2]; } ua;
            ua.hh[0] = *(const bf16x8*)(arow[mt] + kk + 16 * g);
            ua.hh[1] = *(const bf16x8*)(arow[mt] + kk + 16 * g + 8);
            af[mt] = ua.v;
        }
        #pragma unroll
        for (int nt = 0; nt < 4; ++nt)
            bfr[nt] = *(const v16bf*)(brow[nt] + kk + 16 * g);

        if (kk + 64 < K) {                       // prefetch A stream (global_prefetch_b8)
            __builtin_prefetch(arow[0] + kk + 64, 0, 0);
            __builtin_prefetch(arow[1] + kk + 64, 0, 0);
        }

        #pragma unroll
        for (int mt = 0; mt < 2; ++mt)
            #pragma unroll
            for (int nt = 0; nt < 4; ++nt)
                acc[mt][nt] = __builtin_amdgcn_wmma_f32_16x16x32_bf16(
                    false, af[mt], false, bfr[nt], (short)0, acc[mt][nt], false, false);
    }

    #pragma unroll
    for (int mt = 0; mt < 2; ++mt) {
        #pragma unroll
        for (int nt = 0; nt < 4; ++nt) {
            const int col = n0 + nt * 16 + l15;
            const float bv = bias ? bias[col] : 0.0f;
            #pragma unroll
            for (int r = 0; r < 8; ++r) {
                const int row = m0 + mt * 16 + r + 8 * g;   // C/D layout: VGPR r -> M=r (+8 hi lanes)
                float v = acc[mt][nt][r] + bv;
                if (relu) v = v > 0.0f ? v : 0.0f;
                if (Cf) Cf[(size_t)row * ldC + col] = v;
                else    Ch[(size_t)row * ldC + col] = (bf16)v;
            }
        }
    }
}

// ---------------------------------------------------------------------------
// Attention scores + softmax for one (b,h): block handles 16 query rows x 512
// keys. WMMA q.k^T (K = DK = 64), then scale + mask (+causal), softmax in LDS,
// write bf16 attention weights [B,H,S,S].
// ---------------------------------------------------------------------------
__global__ __launch_bounds__(256) void attn_scores(
    const bf16* __restrict__ Q0, int ldQ, int hsQ,     // q[s,d] = Q0[(b*S+s)*ldQ + h*hsQ + d]
    const bf16* __restrict__ K0, int ldK, int hsK,     // k[s,d] = K0[(b*S+s)*ldK + h*hsK + d]
    const float* __restrict__ maskv,                   // [B,S] additive key mask
    int causal,
    bf16* __restrict__ AW)                             // [B,H,S,S]
{
    __shared__ float sc[16][512];                      // 32 KB
    const int bh = blockIdx.y, b = bh >> 3, h = bh & 7;
    const int q0 = blockIdx.x * 16;
    const int lane = threadIdx.x & 31, wave = threadIdx.x >> 5;
    const int l15 = lane & 15, g = lane >> 4;
    const float inv_scale = 0.125f;                    // 1/sqrt(64)

    const bf16* qrow = Q0 + ((size_t)(b * S_) + q0 + l15) * ldQ + h * hsQ;
    v8f acc[4] = {};
    for (int kk = 0; kk < DK_; kk += 32) {
        union { v16bf v; bf16x8 hh[2]; } ua;
        ua.hh[0] = *(const bf16x8*)(qrow + kk + 16 * g);
        ua.hh[1] = *(const bf16x8*)(qrow + kk + 16 * g + 8);
        #pragma unroll
        for (int nt = 0; nt < 4; ++nt) {
            const int n = wave * 64 + nt * 16 + l15;
            const bf16* krow = K0 + ((size_t)(b * S_) + n) * ldK + h * hsK;
            v16bf bfr = *(const v16bf*)(krow + kk + 16 * g);
            acc[nt] = __builtin_amdgcn_wmma_f32_16x16x32_bf16(false, ua.v, false, bfr, (short)0, acc[nt], false, false);
        }
    }
    #pragma unroll
    for (int nt = 0; nt < 4; ++nt) {
        const int n = wave * 64 + nt * 16 + l15;
        const float mk = maskv[b * S_ + n];
        #pragma unroll
        for (int r = 0; r < 8; ++r) {
            const int qr = r + 8 * g;
            float v = acc[nt][r] + mk;
            if (causal && n > (q0 + qr)) v = -__builtin_inff();
            sc[qr][n] = v * inv_scale;
        }
    }
    __syncthreads();

    // softmax: each wave handles 2 rows, lane-strided + shfl_xor reductions
    for (int rr = 0; rr < 2; ++rr) {
        const int row = wave * 2 + rr;
        float mx = -__builtin_inff();
        for (int j = lane; j < 512; j += 32) mx = fmaxf(mx, sc[row][j]);
        for (int off = 16; off > 0; off >>= 1) mx = fmaxf(mx, __shfl_xor(mx, off));
        float sum = 0.0f;
        for (int j = lane; j < 512; j += 32) sum += __expf(sc[row][j] - mx);
        for (int off = 16; off > 0; off >>= 1) sum += __shfl_xor(sum, off);
        const float inv = 1.0f / sum;
        bf16* out = AW + (((size_t)(b * HEADS_ + h)) * S_ + q0 + row) * S_;
        for (int j = lane; j < 512; j += 32)
            out[j] = (bf16)(__expf(sc[row][j] - mx) * inv);
    }
}

// ---------------------------------------------------------------------------
// ctx = attn_w @ V for one (b,h): M=512 rows of weights, N=64, K=512.
// A (attn weights, bf16, contiguous) is WMMA-native; V fragments gathered.
// Writes fp32 ctx at columns [h*64, h*64+64) of [B*S, EMB].
// ---------------------------------------------------------------------------
__global__ __launch_bounds__(256) void attn_ctx(
    const bf16* __restrict__ AW,                       // [B,H,S,S]
    const bf16* __restrict__ V0, int ldV, int hsV,     // v[s,d] = V0[(b*S+s)*ldV + h*hsV + d]
    float* __restrict__ C)                             // [B*S, EMB]
{
    const int bh = blockIdx.y, b = bh >> 3, h = bh & 7;
    const int lane = threadIdx.x & 31, wave = threadIdx.x >> 5;
    const int l15 = lane & 15, g = lane >> 4;
    const int m0 = blockIdx.x * 128 + wave * 16;

    const bf16* arow  = AW + (((size_t)(b * HEADS_ + h)) * S_ + m0 + l15) * S_;
    const bf16* vbase = V0 + (size_t)(b * S_) * ldV + h * hsV;

    v8f acc[4] = {};
    for (int kk = 0; kk < S_; kk += 32) {
        union { v16bf v; bf16x8 hh[2]; } ua;
        ua.hh[0] = *(const bf16x8*)(arow + kk + 8 * g);
        ua.hh[1] = *(const bf16x8*)(arow + kk + 8 * g + 16);
        #pragma unroll
        for (int nt = 0; nt < 4; ++nt) {
            const int d = nt * 16 + l15;
            v16bf bfr;
            #pragma unroll
            for (int e = 0; e < 16; ++e)
                bfr[e] = vbase[(size_t)(kk + 8 * g + (e & 7) + ((e >> 3) << 4)) * ldV + d];
            acc[nt] = __builtin_amdgcn_wmma_f32_16x16x32_bf16(false, ua.v, false, bfr, (short)0, acc[nt], false, false);
        }
    }
    #pragma unroll
    for (int nt = 0; nt < 4; ++nt) {
        const int d = nt * 16 + l15;
        #pragma unroll
        for (int r = 0; r < 8; ++r) {
            const int row = m0 + r + 8 * g;
            C[(size_t)(b * S_ + row) * EMB_ + h * DK_ + d] = acc[nt][r];
        }
    }
}

// ---------------------------------------------------------------------------
// residual add + layernorm over EMB=512; writes fp32 x and bf16 copy.
// ---------------------------------------------------------------------------
__global__ __launch_bounds__(256) void add_ln(
    const float* __restrict__ X, const float* __restrict__ Y,
    const float* __restrict__ gamma, const float* __restrict__ beta,
    float* __restrict__ Xout, bf16* __restrict__ Xbf)
{
    __shared__ float red[256];
    const int row = blockIdx.x, t = threadIdx.x;
    const size_t base = (size_t)row * EMB_;
    float v0 = X[base + t]       + Y[base + t];
    float v1 = X[base + 256 + t] + Y[base + 256 + t];
    red[t] = v0 + v1; __syncthreads();
    for (int s = 128; s > 0; s >>= 1) { if (t < s) red[t] += red[t + s]; __syncthreads(); }
    const float mean = red[0] * (1.0f / 512.0f); __syncthreads();
    const float d0 = v0 - mean, d1 = v1 - mean;
    red[t] = d0 * d0 + d1 * d1; __syncthreads();
    for (int s = 128; s > 0; s >>= 1) { if (t < s) red[t] += red[t + s]; __syncthreads(); }
    const float inv = rsqrtf(red[0] * (1.0f / 512.0f) + 1e-5f);
    const float o0 = d0 * inv * gamma[t]       + beta[t];
    const float o1 = d1 * inv * gamma[256 + t] + beta[256 + t];
    Xout[base + t] = o0;       Xout[base + 256 + t] = o1;
    Xbf[base + t]  = (bf16)o0; Xbf[base + 256 + t]  = (bf16)o1;
}

// ---------------------------------------------------------------------------
// misc elementwise kernels
// ---------------------------------------------------------------------------
__global__ void cvt_f32_bf16(const float* __restrict__ in, bf16* __restrict__ out, int n)
{
    int i = blockIdx.x * blockDim.x + threadIdx.x;
    if (i < n) out[i] = (bf16)in[i];
}

__global__ __launch_bounds__(256) void embed_pos(
    const int* __restrict__ tok, const float* __restrict__ table,
    float* __restrict__ X, bf16* __restrict__ Xbf)
{
    const int row = blockIdx.x;          // b*S + s
    const int s = row & (S_ - 1);
    const int id = tok[row];
    for (int j = threadIdx.x; j < EMB_; j += 256) {
        const float ang = (float)s / __powf(10000.0f, 2.0f * (float)j / (float)EMB_);
        const float pe = (j & 1) ? __cosf(ang) : __sinf(ang);
        const float v = table[(size_t)id * EMB_ + j] + pe;
        X[(size_t)row * EMB_ + j]   = v;
        Xbf[(size_t)row * EMB_ + j] = (bf16)v;
    }
}

__global__ __launch_bounds__(256) void softmax_rows(float* __restrict__ X, int N)
{
    __shared__ float red[256];
    float* p = X + (size_t)blockIdx.x * N;
    const int t = threadIdx.x;
    float mx = -__builtin_inff();
    for (int j = t; j < N; j += 256) mx = fmaxf(mx, p[j]);
    red[t] = mx; __syncthreads();
    for (int s = 128; s > 0; s >>= 1) { if (t < s) red[t] = fmaxf(red[t], red[t + s]); __syncthreads(); }
    mx = red[0]; __syncthreads();
    float sum = 0.0f;
    for (int j = t; j < N; j += 256) sum += __expf(p[j] - mx);
    red[t] = sum; __syncthreads();
    for (int s = 128; s > 0; s >>= 1) { if (t < s) red[t] += red[t + s]; __syncthreads(); }
    const float inv = 1.0f / red[0];
    for (int j = t; j < N; j += 256) p[j] = __expf(p[j] - mx) * inv;
}

// ---------------------------------------------------------------------------
// host orchestration
// ---------------------------------------------------------------------------
extern "C" void kernel_launch(void* const* d_in, const int* in_sizes, int n_in,
                              void* d_out, int out_size, void* d_ws, size_t ws_size,
                              hipStream_t stream)
{
    (void)in_sizes; (void)n_in; (void)out_size; (void)ws_size;

    const int*   enc      = (const int*)  d_in[0];
    const int*   dec      = (const int*)  d_in[1];
    const float* enc_mask = (const float*)d_in[2];
    const float* dec_mask = (const float*)d_in[3];
    // d_in[4], d_in[5] = enc_num/dec_num == 6 (compile-time constant in reference)
    const float* emb    = (const float*)d_in[6];
    const float* w_qkv  = (const float*)d_in[7];
    const float* w_lin  = (const float*)d_in[8];
    const float* b_lin  = (const float*)d_in[9];
    const float* w_ff1  = (const float*)d_in[10];
    const float* b_ff1  = (const float*)d_in[11];
    const float* w_ff2  = (const float*)d_in[12];
    const float* b_ff2  = (const float*)d_in[13];
    const float* w_out  = (const float*)d_in[14];
    const float* b_out  = (const float*)d_in[15];
    const float* g_e1 = (const float*)d_in[16]; const float* be_e1 = (const float*)d_in[17];
    const float* g_e2 = (const float*)d_in[18]; const float* be_e2 = (const float*)d_in[19];
    const float* g_d1 = (const float*)d_in[20]; const float* be_d1 = (const float*)d_in[21];
    const float* g_d2 = (const float*)d_in[22]; const float* be_d2 = (const float*)d_in[23];
    const float* g_d3 = (const float*)d_in[24]; const float* be_d3 = (const float*)d_in[25];

    // workspace carve-out (~150 MB total)
    size_t off = 0;
    auto carve = [&](size_t bytes) {
        void* p = (char*)d_ws + off;
        off = (off + bytes + 255) & ~(size_t)255;
        return p;
    };
    bf16*  wqkv_bf  = (bf16*) carve((size_t)3 * EMB_ * EMB_ * 2);
    bf16*  wlin_bf  = (bf16*) carve((size_t)EMB_ * EMB_ * 2);
    bf16*  wff1_bf  = (bf16*) carve((size_t)FF_ * EMB_ * 2);
    bf16*  wff2_bf  = (bf16*) carve((size_t)EMB_ * FF_ * 2);
    bf16*  wout_bf  = (bf16*) carve((size_t)VOCAB_ * EMB_ * 2);
    float* x_e      = (float*)carve((size_t)NROWS * EMB_ * 4);
    float* x_d      = (float*)carve((size_t)NROWS * EMB_ * 4);
    bf16*  xa_bf    = (bf16*) carve((size_t)NROWS * EMB_ * 2);      // bf16 of current activation
    bf16*  h_bf     = (bf16*) carve((size_t)NROWS * FF_ * 2);       // relu(ff1) bf16
    bf16*  qkv_bf   = (bf16*) carve((size_t)NROWS * 3 * EMB_ * 2);  // per-layer qkv
    bf16*  enckv_bf = (bf16*) carve((size_t)NROWS * 3 * EMB_ * 2);  // encoder qkv for cross-attn
    bf16*  dq_bf    = (bf16*) carve((size_t)NROWS * EMB_ * 2);      // decoder cross-attn q
    bf16*  attnw    = (bf16*) carve((size_t)B_ * HEADS_ * S_ * S_ * 2);
    float* ctx      = (float*)carve((size_t)NROWS * EMB_ * 4);
    float* tmp      = (float*)carve((size_t)NROWS * EMB_ * 4);

    auto gemm = [&](const bf16* A, int ldA, const bf16* W, int ldW, int N, int K,
                    float* Cf, bf16* Ch, int ldC, const float* bias, int relu) {
        gemm_bf16_nt<<<dim3(NROWS / 64, N / 256), 256, 0, stream>>>(
            A, ldA, W, ldW, Cf, Ch, ldC, bias, relu, K);
    };
    auto cvt = [&](const float* src, bf16* dst, int n) {
        cvt_f32_bf16<<<(n + 255) / 256, 256, 0, stream>>>(src, dst, n);
    };

    // weights -> bf16 (every call; deterministic, no caching)
    cvt(w_qkv, wqkv_bf, 3 * EMB_ * EMB_);
    cvt(w_lin, wlin_bf, EMB_ * EMB_);
    cvt(w_ff1, wff1_bf, FF_ * EMB_);
    cvt(w_ff2, wff2_bf, EMB_ * FF_);
    cvt(w_out, wout_bf, VOCAB_ * EMB_);

    const int QKV = 3 * EMB_;      // 1536, head stride 192, q/k/v offsets 0/64/128
    const dim3 SCG(S_ / 16, B_ * HEADS_), CXG(S_ / 128, B_ * HEADS_);

    // ---------------- encoder ----------------
    embed_pos<<<NROWS, 256, 0, stream>>>(enc, emb, x_e, xa_bf);
    for (int l = 0; l < 6; ++l) {
        gemm(xa_bf, EMB_, wqkv_bf, EMB_, QKV, EMB_, nullptr, qkv_bf, QKV, nullptr, 0);
        attn_scores<<<SCG, 256, 0, stream>>>(qkv_bf, QKV, 192, qkv_bf + 64, QKV, 192, enc_mask, 0, attnw);
        attn_ctx<<<CXG, 256, 0, stream>>>(attnw, qkv_bf + 128, QKV, 192, ctx);
        cvt(ctx, xa_bf, NROWS * EMB_);
        gemm(xa_bf, EMB_, wlin_bf, EMB_, EMB_, EMB_, tmp, nullptr, EMB_, b_lin, 0);
        add_ln<<<NROWS, 256, 0, stream>>>(x_e, tmp, g_e1, be_e1, x_e, xa_bf);
        gemm(xa_bf, EMB_, wff1_bf, EMB_, FF_, EMB_, nullptr, h_bf, FF_, b_ff1, 1);
        gemm(h_bf, FF_, wff2_bf, FF_, EMB_, FF_, tmp, nullptr, EMB_, b_ff2, 0);
        add_ln<<<NROWS, 256, 0, stream>>>(x_e, tmp, g_e2, be_e2, x_e, xa_bf);
    }
    // encoder k/v for cross attention (full qkv of final x_e)
    gemm(xa_bf, EMB_, wqkv_bf, EMB_, QKV, EMB_, nullptr, enckv_bf, QKV, nullptr, 0);

    // ---------------- decoder ----------------
    embed_pos<<<NROWS, 256, 0, stream>>>(dec, emb, x_d, xa_bf);
    for (int l = 0; l < 6; ++l) {
        // masked self-attention
        gemm(xa_bf, EMB_, wqkv_bf, EMB_, QKV, EMB_, nullptr, qkv_bf, QKV, nullptr, 0);
        attn_scores<<<SCG, 256, 0, stream>>>(qkv_bf, QKV, 192, qkv_bf + 64, QKV, 192, dec_mask, 1, attnw);
        attn_ctx<<<CXG, 256, 0, stream>>>(attnw, qkv_bf + 128, QKV, 192, ctx);
        cvt(ctx, xa_bf, NROWS * EMB_);
        gemm(xa_bf, EMB_, wlin_bf, EMB_, EMB_, EMB_, tmp, nullptr, EMB_, b_lin, 0);
        add_ln<<<NROWS, 256, 0, stream>>>(x_d, tmp, g_d1, be_d1, x_d, xa_bf);
        // cross-attention: q = first EMB cols of x_d @ w_qkv^T (cols h*64+d)
        gemm(xa_bf, EMB_, wqkv_bf, EMB_, EMB_, EMB_, nullptr, dq_bf, EMB_, nullptr, 0);
        attn_scores<<<SCG, 256, 0, stream>>>(dq_bf, EMB_, DK_, enckv_bf + 64, QKV, 192, enc_mask, 0, attnw);
        attn_ctx<<<CXG, 256, 0, stream>>>(attnw, enckv_bf + 128, QKV, 192, ctx);
        cvt(ctx, xa_bf, NROWS * EMB_);
        gemm(xa_bf, EMB_, wlin_bf, EMB_, EMB_, EMB_, tmp, nullptr, EMB_, b_lin, 0);
        add_ln<<<NROWS, 256, 0, stream>>>(x_d, tmp, g_d2, be_d2, x_d, xa_bf);
        // ffn
        gemm(xa_bf, EMB_, wff1_bf, EMB_, FF_, EMB_, nullptr, h_bf, FF_, b_ff1, 1);
        gemm(h_bf, FF_, wff2_bf, FF_, EMB_, FF_, tmp, nullptr, EMB_, b_ff2, 0);
        add_ln<<<NROWS, 256, 0, stream>>>(x_d, tmp, g_d3, be_d3, x_d, xa_bf);
    }

    // vocab projection + softmax into d_out
    gemm(xa_bf, EMB_, wout_bf, EMB_, VOCAB_, EMB_, (float*)d_out, nullptr, VOCAB_, b_out, 0);
    softmax_rows<<<NROWS, 256, 0, stream>>>((float*)d_out, VOCAB_);
}
